// LSTM_66176856096910
// MI455X (gfx1250) — matrix operand order repaired and checked by reading, not compiled, not run on hardware
//
#include <hip/hip_runtime.h>
#include <math.h>

#define Bsz 32
#define Tsz 256
#define Dsz 512
#define Hsz 512
#define Csz 4096
#define OUTC 1536

typedef unsigned short u16;
typedef unsigned int   u32;
typedef __attribute__((ext_vector_type(16))) __bf16 v16bf;
typedef __attribute__((ext_vector_type(8)))  float  v8f;
typedef __attribute__((ext_vector_type(4)))  u32    v4u;
typedef __attribute__((ext_vector_type(8)))  u32    v8u;

union Frag {
  v16bf v;
  v4u   q[2];
  v8u   w;
};

__device__ __forceinline__ u16 f32_to_bf16(float x) {
  union { float f; u32 u; } cv; cv.f = x;
  u32 u = cv.u;
  u32 r = (u + 0x7FFFu + ((u >> 16) & 1u)) >> 16;  // round to nearest even
  return (u16)r;
}

__device__ __forceinline__ float sigmoidf_(float x) {
  return 1.0f / (1.0f + __expf(-x));
}

// A-fragment (16-bit A 16x32 MxK layout): lane holds row m = lane&15,
// K chunks [k0 + half*8, +8) and [k0+16 + half*8, +8)   (half = lane>>4)
__device__ __forceinline__ v16bf load_a32(const u16* rowk0, int half) {
  Frag f;
  f.q[0] = *(const v4u*)(rowk0 + half * 8);
  f.q[1] = *(const v4u*)(rowk0 + 16 + half * 8);
  return f.v;
}

// B-fragment (16-bit B 32x16 KxN): lane holds column n = lane&15,
// K chunks [k0 + half*8, +8) and [k0+16 + half*8, +8) (same as A). Weight is [N,K]
// row-major, so these are two 16-byte loads from the weight row.
__device__ __forceinline__ v16bf load_b32(const u16* rowk0, int half) {
  Frag f;
  f.q[0] = *(const v4u*)(rowk0 + half * 8);
  f.q[1] = *(const v4u*)(rowk0 + 16 + half * 8);
  return f.v;
}

__global__ void cvt_f32_bf16(const float* __restrict__ in, u16* __restrict__ out, int n) {
  int i = blockIdx.x * blockDim.x + threadIdx.x;
  if (i < n) out[i] = f32_to_bf16(in[i]);
}

// One timestep, both directions (blockIdx.y = dir).
// gates = [h_prev, x_t] @ [W_h, W_in]^T + bias  (fused K = H + D = 1024)
// Each wave owns one 16-wide cc tile of C and all 4 gates for all 32 batch
// rows -> cell nonlinearity fully in-register, writes c (f32) and
// act = sigmoid(o)*tanh(c_new) (bf16) for the projection GEMM.
__global__ __launch_bounds__(256) void lstm_cell_gates(
    const u16* __restrict__ xbf,    // [B,T,D] bf16
    const u16* __restrict__ hbf,    // [2,B,H] bf16
    const u16* __restrict__ Whbf,   // [2,4C,H] bf16
    const u16* __restrict__ Winbf,  // [2,4C,D] bf16
    const float* __restrict__ bias_f,
    const float* __restrict__ bias_b,
    const int* __restrict__ lengths,
    float* __restrict__ c_state,    // [2,B,C] f32
    u16* __restrict__ actbf,        // [2,B,C] bf16
    int t) {
  const int wave = threadIdx.x >> 5;
  const int lane = threadIdx.x & 31;
  const int dir  = blockIdx.y;
  const int ccTile = blockIdx.x * 8 + wave;   // 0..255 (C/16)
  const int t_eff = dir ? (Tsz - 1 - t) : t;
  const int mrow = lane & 15;
  const int half = lane >> 4;

  const float* bias = dir ? bias_b : bias_f;
  const u16* Wh    = Whbf  + (size_t)dir * 4 * Csz * Hsz;
  const u16* Win   = Winbf + (size_t)dir * 4 * Csz * Dsz;
  const u16* hrow0 = hbf   + (size_t)dir * Bsz * Hsz;

  v8f acc[2][4] = {};   // [m-tile][gate]

  // recurrent half: K over H, A rows = h_prev
  for (int k0 = 0; k0 < Hsz; k0 += 32) {
    v16bf a0 = load_a32(hrow0 + (size_t)mrow * Hsz + k0, half);
    v16bf a1 = load_a32(hrow0 + (size_t)(16 + mrow) * Hsz + k0, half);
#pragma unroll
    for (int g = 0; g < 4; ++g) {
      const int nrow = g * Csz + ccTile * 16 + mrow;   // n == lane&15 == mrow
      v16bf bb = load_b32(Wh + (size_t)nrow * Hsz + k0, half);
      acc[0][g] = __builtin_amdgcn_wmma_f32_16x16x32_bf16(false, a0, false, bb, (short)0, acc[0][g], false, false);
      acc[1][g] = __builtin_amdgcn_wmma_f32_16x16x32_bf16(false, a1, false, bb, (short)0, acc[1][g], false, false);
    }
  }
  // input half: K over D, A rows = x[b][t_eff][:]
  for (int k0 = 0; k0 < Dsz; k0 += 32) {
    v16bf a0 = load_a32(xbf + ((size_t)mrow * Tsz + t_eff) * Dsz + k0, half);
    v16bf a1 = load_a32(xbf + ((size_t)(16 + mrow) * Tsz + t_eff) * Dsz + k0, half);
#pragma unroll
    for (int g = 0; g < 4; ++g) {
      const int nrow = g * Csz + ccTile * 16 + mrow;
      v16bf bb = load_b32(Win + (size_t)nrow * Dsz + k0, half);
      acc[0][g] = __builtin_amdgcn_wmma_f32_16x16x32_bf16(false, a0, false, bb, (short)0, acc[0][g], false, false);
      acc[1][g] = __builtin_amdgcn_wmma_f32_16x16x32_bf16(false, a1, false, bb, (short)0, acc[1][g], false, false);
    }
  }

  // cell epilogue. D layout: element v -> row m = v + 8*half, col n = lane&15
  const int cc = ccTile * 16 + mrow;
  const float bi  = bias[0 * Csz + cc];
  const float bfv = bias[1 * Csz + cc];
  const float bg  = bias[2 * Csz + cc];
  const float bo  = bias[3 * Csz + cc];
#pragma unroll
  for (int mi = 0; mi < 2; ++mi) {
#pragma unroll
    for (int v = 0; v < 8; ++v) {
      const int b = mi * 16 + v + 8 * half;
      const size_t idx = ((size_t)dir * Bsz + b) * Csz + cc;
      const float gi = acc[mi][0][v] + bi;
      const float gf = acc[mi][1][v] + bfv;
      const float gg = acc[mi][2][v] + bg;
      const float go = acc[mi][3][v] + bo;
      const float cprev = c_state[idx];
      const float cnew  = sigmoidf_(gi) * tanhf(gg) + sigmoidf_(gf) * cprev;
      const bool  msk   = t_eff < lengths[b];
      c_state[idx] = msk ? cnew : cprev;
      actbf[idx]   = f32_to_bf16(sigmoidf_(go) * tanhf(cnew));
    }
  }
}

// h_new = act @ W_p^T ; masked h update; write masked output slice.
__global__ __launch_bounds__(256) void lstm_proj(
    const u16* __restrict__ actbf,  // [2,B,C] bf16
    const u16* __restrict__ Wpbf,   // [2,H,C] bf16
    const int* __restrict__ lengths,
    float* __restrict__ h_state,    // [2,B,H] f32
    u16* __restrict__ hbf,          // [2,B,H] bf16
    float* __restrict__ out,        // [B,T,1536]
    int t) {
  const int wave = threadIdx.x >> 5;
  const int lane = threadIdx.x & 31;
  const int dir  = blockIdx.y;
  const int nTile = blockIdx.x * 8 + wave;   // 0..31 (H/16)
  const int t_eff = dir ? (Tsz - 1 - t) : t;
  const int mrow = lane & 15;
  const int half = lane >> 4;

  const u16* A  = actbf + (size_t)dir * Bsz * Csz;
  const u16* Wp = Wpbf  + (size_t)dir * Hsz * Csz;

  v8f acc[2] = {};
  for (int k0 = 0; k0 < Csz; k0 += 32) {
    v16bf a0 = load_a32(A + (size_t)mrow * Csz + k0, half);
    v16bf a1 = load_a32(A + (size_t)(16 + mrow) * Csz + k0, half);
    v16bf bb = load_b32(Wp + (size_t)(nTile * 16 + mrow) * Csz + k0, half);
    acc[0] = __builtin_amdgcn_wmma_f32_16x16x32_bf16(false, a0, false, bb, (short)0, acc[0], false, false);
    acc[1] = __builtin_amdgcn_wmma_f32_16x16x32_bf16(false, a1, false, bb, (short)0, acc[1], false, false);
  }

  const int hcol = nTile * 16 + mrow;
#pragma unroll
  for (int mi = 0; mi < 2; ++mi) {
#pragma unroll
    for (int v = 0; v < 8; ++v) {
      const int b = mi * 16 + v + 8 * half;
      const bool msk = t_eff < lengths[b];
      const size_t hidx = ((size_t)dir * Bsz + b) * Hsz + hcol;
      const float hnew  = acc[mi][v];
      const float hprev = h_state[hidx];
      const float hw    = msk ? hnew : hprev;
      h_state[hidx] = hw;
      hbf[hidx]     = f32_to_bf16(hw);
      out[((size_t)b * Tsz + t_eff) * OUTC + dir * Hsz + hcol] = msk ? hnew : 0.0f;
    }
  }
}

__global__ void embed_copy(const float* __restrict__ e, float* __restrict__ out, int n) {
  int i = blockIdx.x * blockDim.x + threadIdx.x;
  if (i < n) {
    int col = i % Hsz;
    int bt  = i / Hsz;
    out[(size_t)bt * OUTC + 2 * Hsz + col] = e[i];
  }
}

extern "C" void kernel_launch(void* const* d_in, const int* in_sizes, int n_in,
                              void* d_out, int out_size, void* d_ws, size_t ws_size,
                              hipStream_t stream) {
  (void)in_sizes; (void)n_in; (void)out_size; (void)ws_size;
  const float* batch_embed = (const float*)d_in[0];
  const float* input_layer = (const float*)d_in[1];
  const int*   batch_len   = (const int*)d_in[2];
  const float* W_in_f = (const float*)d_in[3];
  const float* W_h_f  = (const float*)d_in[4];
  const float* b_f    = (const float*)d_in[5];
  const float* W_p_f  = (const float*)d_in[6];
  const float* W_in_b = (const float*)d_in[7];
  const float* W_h_b  = (const float*)d_in[8];
  const float* b_b    = (const float*)d_in[9];
  const float* W_p_b  = (const float*)d_in[10];
  float* out = (float*)d_out;

  char* ws = (char*)d_ws;
  size_t off = 0;
  auto alloc = [&](size_t bytes) -> void* {
    void* p = ws + off;
    off = (off + bytes + 255) & ~(size_t)255;
    return p;
  };
  u16*   xbf   = (u16*)alloc((size_t)Bsz * Tsz * Dsz * 2);        //  8 MB
  u16*   Winbf = (u16*)alloc((size_t)2 * 4 * Csz * Dsz * 2);      // 32 MB
  u16*   Whbf  = (u16*)alloc((size_t)2 * 4 * Csz * Hsz * 2);      // 32 MB
  u16*   Wpbf  = (u16*)alloc((size_t)2 * Hsz * Csz * 2);          //  8 MB
  float* hst   = (float*)alloc((size_t)2 * Bsz * Hsz * 4);
  u16*   hbf   = (u16*)alloc((size_t)2 * Bsz * Hsz * 2);
  float* cst   = (float*)alloc((size_t)2 * Bsz * Csz * 4);
  u16*   actbf = (u16*)alloc((size_t)2 * Bsz * Csz * 2);

  auto cvt = [&](const float* src, u16* dst, int n) {
    cvt_f32_bf16<<<(n + 255) / 256, 256, 0, stream>>>(src, dst, n);
  };
  cvt(input_layer, xbf, Bsz * Tsz * Dsz);
  cvt(W_in_f, Winbf, 4 * Csz * Dsz);
  cvt(W_in_b, Winbf + (size_t)4 * Csz * Dsz, 4 * Csz * Dsz);
  cvt(W_h_f, Whbf, 4 * Csz * Hsz);
  cvt(W_h_b, Whbf + (size_t)4 * Csz * Hsz, 4 * Csz * Hsz);
  cvt(W_p_f, Wpbf, Hsz * Csz);
  cvt(W_p_b, Wpbf + (size_t)Hsz * Csz, Hsz * Csz);

  hipMemsetAsync(hst, 0, (size_t)2 * Bsz * Hsz * 4, stream);
  hipMemsetAsync(hbf, 0, (size_t)2 * Bsz * Hsz * 2, stream);
  hipMemsetAsync(cst, 0, (size_t)2 * Bsz * Csz * 4, stream);

  for (int t = 0; t < Tsz; ++t) {
    lstm_cell_gates<<<dim3(32, 2), 256, 0, stream>>>(
        xbf, hbf, Whbf, Winbf, b_f, b_b, batch_len, cst, actbf, t);
    lstm_proj<<<dim3(4, 2), 256, 0, stream>>>(
        actbf, Wpbf, batch_len, hst, hbf, out, t);
  }

  const int ne = Bsz * Tsz * Hsz;
  embed_copy<<<(ne + 255) / 256, 256, 0, stream>>>(batch_embed, out, ne);
}
